// CPImplicitPairwise_23252952941220
// MI455X (gfx1250) — matrix operand order repaired and checked by reading, not compiled, not run on hardware
//
#include <hip/hip_runtime.h>
#include <hip/hip_bf16.h>
#include <math.h>

typedef __attribute__((ext_vector_type(16))) _Float16 v16h;
typedef __attribute__((ext_vector_type(8)))  float    v8f;
typedef __attribute__((ext_vector_type(2)))  float    v2f;
typedef __attribute__((ext_vector_type(4)))  unsigned int u32x4;
typedef __attribute__((ext_vector_type(8)))  int      i32x8;
typedef __attribute__((ext_vector_type(4)))  int      i32x4;

#define BSZ   2
#define CCH   128
#define FF    16200
#define NVOX  2048
#define KSEL  256
#define NREL  3

// ---------------- K0: positional encoding + xi ----------------
__global__ void k_pexi(const float* __restrict__ input, const int* __restrict__ masks,
                       float* __restrict__ pe_m, float* __restrict__ xi) {
    int idx = blockIdx.x * blockDim.x + threadIdx.x;       // BSZ*NVOX*CCH
    if (idx >= BSZ * NVOX * CCH) return;
    int c  = idx & (CCH - 1);
    int bn = idx >> 7;
    int b  = bn >> 11;
    int pos = masks[bn];
    int i2 = c >> 1;
    float div = __expf(-(float)(2 * i2) * (9.210340371976184f / 128.0f));
    float ang = (float)pos * div;
    float pe  = (c & 1) ? __cosf(ang) : __sinf(ang);
    pe_m[idx] = pe;
    xi[idx]   = input[((size_t)b * CCH + c) * FF + pos] + pe;
}

// ---------------- K0b: pack W_p1 / W_p2 into WMMA B-fragment layout (f16) ----
// B 32x16 packed here in contiguous k order (A uses the same): lane l (n = l&15, hi = l>>4), half j in 0..15: K = hi*16 + j
__global__ void k_wfrag(const float* __restrict__ W_p1, const float* __restrict__ W_p2,
                        _Float16* __restrict__ wb1, _Float16* __restrict__ wb2) {
    int t = threadIdx.x;
    for (int e = t; e < 8 * 512; e += 256) {               // W_p1 [128,32]: kc 0..3, nt 0..1
        int tile = e >> 9, kc = tile >> 1, nt = tile & 1;
        int r = e & 511, lane = r >> 4, j = r & 15;
        int kk = kc * 32 + (lane >> 4) * 16 + j;
        int nn = nt * 16 + (lane & 15);
        wb1[e] = (_Float16)W_p1[kk * 32 + nn];
    }
    for (int e = t; e < 2 * 512; e += 256) {               // W_p2 [32,32]: nt 0..1
        int tile = e >> 9, nt = tile & 1;
        int r = e & 511, lane = r >> 4, j = r & 15;
        int kk = (lane >> 4) * 16 + j;
        int nn = nt * 16 + (lane & 15);
        wb2[e] = (_Float16)W_p2[kk * 32 + nn];
    }
}

// ---------------- K0c: a = xi@W_r1 ; b = pe_m@W_r1 + b_r1 ----------------
__global__ void k_ab(const float* __restrict__ xi, const float* __restrict__ pe_m,
                     const float* __restrict__ W_r1, const float* __restrict__ b_r1,
                     float* __restrict__ a_ws, float* __restrict__ b_ws) {
    int idx = blockIdx.x * blockDim.x + threadIdx.x;       // BSZ*NVOX*32
    if (idx >= BSZ * NVOX * 32) return;
    int j = idx & 31, bn = idx >> 5;
    float sa = 0.f, sb = 0.f;
    const float* xr = &xi[(size_t)bn * CCH];
    const float* pr = &pe_m[(size_t)bn * CCH];
    for (int c = 0; c < CCH; c++) {
        float w = W_r1[c * 32 + j];
        sa += xr[c] * w;
        sb += pr[c] * w;
    }
    a_ws[idx] = sa;
    b_ws[idx] = sb + b_r1[j];
}

// ---------------- K1: pairwise relate logits + per-row top-K (bitonic) ------
__global__ void k_relate_topk(const float* __restrict__ a_ws, const float* __restrict__ b_ws,
                              const float* __restrict__ g_r1, const float* __restrict__ be_r1,
                              const float* __restrict__ W_r2, const float* __restrict__ b_r2,
                              float* __restrict__ topv, int* __restrict__ topi_out) {
    __shared__ float s_val[NVOX];
    __shared__ int   s_idx[NVOX];
    __shared__ float s_a[32], s_g[32], s_be[32], s_w2[32];
    int t  = threadIdx.x;
    int bn = blockIdx.x;
    int b  = bn >> 11;
    if (t < 32) { s_a[t] = a_ws[(size_t)bn * 32 + t]; s_g[t] = g_r1[t]; s_be[t] = be_r1[t]; s_w2[t] = W_r2[t]; }
    __syncthreads();
    float br2 = b_r2[0];
    for (int j = t; j < NVOX; j += 256) {
        const float* bj = &b_ws[((size_t)b * NVOX + j) * 32];
        float acc = 0.f;
        #pragma unroll
        for (int f = 0; f < 32; f++) {
            float h = s_g[f] * (s_a[f] + bj[f]) + s_be[f];
            h = h > 0.f ? h : 0.f;
            acc += h * s_w2[f];
        }
        s_val[j] = 1.0f / (1.0f + __expf(-(acc + br2)));
        s_idx[j] = j;
    }
    __syncthreads();
    // bitonic sort: descending by value, ties broken by ascending index
    for (int k2 = 2; k2 <= NVOX; k2 <<= 1) {
        for (int j2 = k2 >> 1; j2 > 0; j2 >>= 1) {
            for (int e = t; e < NVOX; e += 256) {
                int ixj = e ^ j2;
                if (ixj > e) {
                    float va = s_val[e], vb = s_val[ixj];
                    int   ia = s_idx[e], ib = s_idx[ixj];
                    bool beforeBA = (vb > va) || (vb == va && ib < ia);
                    bool dir = ((e & k2) == 0);
                    if (dir ? beforeBA : !beforeBA) {
                        s_val[e] = vb; s_val[ixj] = va;
                        s_idx[e] = ib; s_idx[ixj] = ia;
                    }
                }
            }
            __syncthreads();
        }
    }
    if (t < KSEL) {
        topv[(size_t)bn * KSEL + t]     = s_val[t];
        topi_out[(size_t)bn * KSEL + t] = s_idx[t];
    }
}

// fragment address helpers for 16-bit A 16x32 (MxK) layout
__device__ __forceinline__ int a_lane(int m, int kk) { return m + (((kk >> 4) & 1) << 4); }
__device__ __forceinline__ int a_j(int kk)           { return kk & 15; }

// ---------------- K2: per-voxel pair MLP via WMMA + softmax + ctx -----------
// one block per (b,n); 256 threads = 8 waves; GEMM1 256x128x32, GEMM2 256x32x32
__global__ void k_pairmlp(const float* __restrict__ xi, const float* __restrict__ pe_m,
                          const int* __restrict__ topi_out, const float* __restrict__ topv_ws,
                          const _Float16* __restrict__ wb1, const _Float16* __restrict__ wb2,
                          const float* __restrict__ g_p1, const float* __restrict__ b_p1, const float* __restrict__ be_p1,
                          const float* __restrict__ g_p2, const float* __restrict__ b_p2, const float* __restrict__ be_p2,
                          const float* __restrict__ W_p3, const float* __restrict__ b_p3,
                          float* __restrict__ P_logits_out, float* __restrict__ ctx_ws) {
    int t = threadIdx.x, bn = blockIdx.x, b = bn >> 11;
    int lane = t & 31, wave = t >> 5;

    __shared__ int   s_topi[KSEL];
    __shared__ float s_topv[KSEL];
    __shared__ float s_xin[CCH];
    __shared__ __align__(32) _Float16 s_zh[16 * 512];     // z chunk, reused for h2
    __shared__ __align__(32) _Float16 s_h1[16 * 512];
    __shared__ __align__(32) _Float16 s_wb1[8 * 512];
    __shared__ __align__(32) _Float16 s_wb2[2 * 512];
    __shared__ float s_P[KSEL * 3];
    __shared__ float s_colsum[3];
    __shared__ float s_c1[96], s_c2[96], s_w3[96], s_b3[3];

    s_topi[t] = topi_out[(size_t)bn * KSEL + t];
    s_topv[t] = topv_ws[(size_t)bn * KSEL + t];
    if (t < CCH) s_xin[t] = xi[(size_t)bn * CCH + t];
    for (int e = t; e < 8 * 512; e += 256) s_wb1[e] = wb1[e];
    for (int e = t; e < 2 * 512; e += 256) s_wb2[e] = wb2[e];
    if (t < 32) {
        s_c1[t] = g_p1[t]; s_c1[32 + t] = b_p1[t]; s_c1[64 + t] = be_p1[t];
        s_c2[t] = g_p2[t]; s_c2[32 + t] = b_p2[t]; s_c2[64 + t] = be_p2[t];
    }
    if (t < 96) s_w3[t] = W_p3[t];
    if (t < 3)  { s_b3[t] = b_p3[t]; s_colsum[t] = 0.f; }
    __syncthreads();

    // ---- GEMM1: h1_pre = z @ W_p1 ; z streamed in 4 K-chunks of 32 ----
    v8f acc[2][2] = {};
    for (int kc = 0; kc < 4; kc++) {
        {   // build z chunk (row t, cols kc*32..kc*32+31) in A-fragment layout
            int krow = t, m = krow & 15, tb = (krow >> 4) * 512;
            const float* per = &pe_m[((size_t)b * NVOX + s_topi[krow]) * CCH + kc * 32];
            #pragma unroll
            for (int kk = 0; kk < 32; kk++) {
                float z = s_xin[kc * 32 + kk] + per[kk];
                s_zh[tb + a_lane(m, kk) * 16 + a_j(kk)] = (_Float16)z;
            }
        }
        __syncthreads();
        #pragma unroll
        for (int mi = 0; mi < 2; mi++) {
            int mt = wave * 2 + mi;
            v16h av = *reinterpret_cast<const v16h*>(&s_zh[mt * 512 + lane * 16]);
            #pragma unroll
            for (int nt = 0; nt < 2; nt++) {
                v16h bv = *reinterpret_cast<const v16h*>(&s_wb1[(kc * 2 + nt) * 512 + lane * 16]);
                acc[mi][nt] = __builtin_amdgcn_wmma_f32_16x16x32_f16(
                    false, av, false, bv, (short)0, acc[mi][nt], false, false);
            }
        }
        __syncthreads();
    }
    // h1 epilogue: relu(g1*(x+b1)+be1), repack into A-frag layout for GEMM2
    #pragma unroll
    for (int mi = 0; mi < 2; mi++) {
        int mt = wave * 2 + mi;
        #pragma unroll
        for (int nt = 0; nt < 2; nt++) {
            int col = nt * 16 + (lane & 15);
            float g = s_c1[col], bb = s_c1[32 + col], be = s_c1[64 + col];
            #pragma unroll
            for (int v = 0; v < 8; v++) {
                int M = v + ((lane >> 4) << 3);
                float h = g * (acc[mi][nt][v] + bb) + be;
                h = h > 0.f ? h : 0.f;
                s_h1[mt * 512 + a_lane(M, col) * 16 + a_j(col)] = (_Float16)h;
            }
        }
    }
    __syncthreads();

    // ---- GEMM2: h2_pre = h1 @ W_p2 (single K step) ----
    v8f acc2[2][2] = {};
    #pragma unroll
    for (int mi = 0; mi < 2; mi++) {
        int mt = wave * 2 + mi;
        v16h av = *reinterpret_cast<const v16h*>(&s_h1[mt * 512 + lane * 16]);
        #pragma unroll
        for (int nt = 0; nt < 2; nt++) {
            v16h bv = *reinterpret_cast<const v16h*>(&s_wb2[nt * 512 + lane * 16]);
            acc2[mi][nt] = __builtin_amdgcn_wmma_f32_16x16x32_f16(
                false, av, false, bv, (short)0, acc2[mi][nt], false, false);
        }
    }
    // h2 epilogue into s_zh (reuse)
    #pragma unroll
    for (int mi = 0; mi < 2; mi++) {
        int mt = wave * 2 + mi;
        #pragma unroll
        for (int nt = 0; nt < 2; nt++) {
            int col = nt * 16 + (lane & 15);
            float g = s_c2[col], bb = s_c2[32 + col], be = s_c2[64 + col];
            #pragma unroll
            for (int v = 0; v < 8; v++) {
                int M = v + ((lane >> 4) << 3);
                float h = g * (acc2[mi][nt][v] + bb) + be;
                h = h > 0.f ? h : 0.f;
                s_zh[mt * 512 + a_lane(M, col) * 16 + a_j(col)] = (_Float16)h;
            }
        }
    }
    __syncthreads();

    // ---- P_logit = h2 @ W_p3 + b_p3 ; softmax(3) * topv ----
    {
        int krow = t, m = krow & 15, tb = (krow >> 4) * 512;
        float l0 = s_b3[0], l1 = s_b3[1], l2 = s_b3[2];
        #pragma unroll
        for (int c = 0; c < 32; c++) {
            float h2v = (float)s_zh[tb + a_lane(m, c) * 16 + a_j(c)];
            l0 += h2v * s_w3[c * 3 + 0];
            l1 += h2v * s_w3[c * 3 + 1];
            l2 += h2v * s_w3[c * 3 + 2];
        }
        size_t ob = ((size_t)bn * KSEL + krow) * 3;
        P_logits_out[ob + 0] = l0; P_logits_out[ob + 1] = l1; P_logits_out[ob + 2] = l2;
        float mx = fmaxf(l0, fmaxf(l1, l2));
        float e0 = __expf(l0 - mx), e1 = __expf(l1 - mx), e2 = __expf(l2 - mx);
        float inv = s_topv[krow] / (e0 + e1 + e2);
        float p0 = e0 * inv, p1 = e1 * inv, p2 = e2 * inv;
        s_P[krow * 3 + 0] = p0; s_P[krow * 3 + 1] = p1; s_P[krow * 3 + 2] = p2;
        atomicAdd(&s_colsum[0], p0); atomicAdd(&s_colsum[1], p1); atomicAdd(&s_colsum[2], p2);
    }
    __syncthreads();

    // ---- ctx[n,r,c] = sum_k P[k,r]*feats[k,c] / colsum[r] ----
    {
        int c = t & 127, grp = t >> 7;
        float a0 = 0.f, a1 = 0.f;
        const float* xib = &xi[(size_t)b * NVOX * CCH];
        for (int k = 0; k < KSEL; k++) {
            float f = xib[(size_t)s_topi[k] * CCH + c];
            if (grp == 0) { a0 += s_P[k * 3 + 0] * f; a1 += s_P[k * 3 + 2] * f; }
            else          { a0 += s_P[k * 3 + 1] * f; }
        }
        size_t cb = (size_t)bn * (NREL * CCH);
        if (grp == 0) {
            ctx_ws[cb + 0 * CCH + c] = a0 / s_colsum[0];
            ctx_ws[cb + 2 * CCH + c] = a1 / s_colsum[2];
        } else {
            ctx_ws[cb + 1 * CCH + c] = a0 / s_colsum[1];
        }
    }
}

// ---------------- K3a: dense base conv via f32 WMMA + TDM tile load --------
// D[o(128), f(64)] = W_rs[0:128]^T @ x_tile ; K = 128, v_wmma_f32_16x16x4_f32
__global__ void k_conv_base(const float* __restrict__ input, const float* __restrict__ W_rs,
                            const float* __restrict__ b_rs, float* __restrict__ out) {
    __shared__ __align__(16) float s_x[128 * 64];
    int b = blockIdx.y, f0 = blockIdx.x * 64, t = threadIdx.x;
    int lane = t & 31, wave = t >> 5;
    int fw = FF - f0; if (fw > 64) fw = 64;

#if __has_builtin(__builtin_amdgcn_tensor_load_to_lds)
    // TDM: 2D descriptor, tile 64 x 128 f32, row stride FF; OOB cols read zero
    if (wave == 0) {
        u32x4 g0; i32x8 g1;
        i32x4 g2 = {0, 0, 0, 0}, g3 = {0, 0, 0, 0};
        i32x8 g4 = {0, 0, 0, 0, 0, 0, 0, 0};
        unsigned long long ga = (unsigned long long)(const void*)(input + (size_t)b * CCH * FF + f0);
        unsigned int ldsoff = (unsigned int)(unsigned long long)(const void*)(&s_x[0]);
        int dim0 = FF - f0;                       // remaining cols from tile start
        g0[0] = 1u;                               // count=1 (valid), no gather
        g0[1] = ldsoff;                           // lds_addr
        g0[2] = (unsigned int)ga;                 // global_addr[31:0]
        g0[3] = (unsigned int)((ga >> 32) & 0x01ffffffu) | (2u << 30);  // addr[56:32] | type=2
        g1[0] = (int)(2u << 16);                  // data_size = 4B
        g1[1] = (int)(((unsigned)dim0 & 0xffffu) << 16);              // tensor_dim0[15:0]
        g1[2] = (int)((((unsigned)dim0 >> 16) & 0xffffu) | (128u << 16)); // dim0[31:16] | tensor_dim1 lo
        g1[3] = (int)(64u << 16);                 // tensor_dim1 hi=0 | tile_dim0=64
        g1[4] = 128;                              // tile_dim1=128, tile_dim2=0
        g1[5] = FF;                               // tensor_dim0_stride[31:0]
        g1[6] = 0;                                // stride hi | dim1_stride lo
        g1[7] = 0;
        __builtin_amdgcn_tensor_load_to_lds(g0, g1, g2, g3, g4, 0);
        __builtin_amdgcn_s_wait_tensorcnt(0);
    }
    __syncthreads();
#else
    for (int e = t; e < 128 * 64; e += 256) {
        int i = e >> 6, ft = e & 63;
        s_x[e] = (ft < fw) ? input[((size_t)b * CCH + i) * FF + f0 + ft] : 0.f;
    }
    __syncthreads();
#endif

    // wave w owns M-tile (o = w*16..w*16+15), 4 N-tiles of f
    int hi = lane >> 4, nl = lane & 15;
    int oa = wave * 16 + nl;                      // A-fragment row
    v8f acc[4] = {};
    for (int ks = 0; ks < 32; ks++) {
        int k0 = ks * 4 + hi * 2;                 // f32 A/B: K = (lane>>4)*2 + v
        v2f av;
        av[0] = W_rs[(k0 + 0) * CCH + oa];
        av[1] = W_rs[(k0 + 1) * CCH + oa];
        #pragma unroll
        for (int nt = 0; nt < 4; nt++) {
            v2f bv;
            bv[0] = s_x[(k0 + 0) * 64 + nt * 16 + nl];
            bv[1] = s_x[(k0 + 1) * 64 + nt * 16 + nl];
            acc[nt] = __builtin_amdgcn_wmma_f32_16x16x4_f32(
                false, av, false, bv, (short)0, acc[nt], false, false);
        }
    }
    // epilogue: D layout N = lane&15, M = v + (lane>>4)*8
    #pragma unroll
    for (int nt = 0; nt < 4; nt++) {
        int ft = nt * 16 + nl;
        if (ft < fw) {
            #pragma unroll
            for (int v = 0; v < 8; v++) {
                int orow = wave * 16 + v + hi * 8;
                out[((size_t)b * CCH + orow) * FF + f0 + ft] = acc[nt][v] + b_rs[orow];
            }
        }
    }
}

// ---------------- K3b: sparse context add at masked columns ----------------
__global__ void k_conv_ctx(const int* __restrict__ masks, const float* __restrict__ ctx_ws,
                           const float* __restrict__ W_rs, float* __restrict__ out) {
    __shared__ float s_ctx[NREL * CCH];
    int bn = blockIdx.x, b = bn >> 11, t = threadIdx.x;   // 128 threads, t = o
    for (int e = t; e < NREL * CCH; e += 128) s_ctx[e] = ctx_ws[(size_t)bn * (NREL * CCH) + e];
    __syncthreads();
    int f = masks[bn];
    float acc = 0.f;
    for (int row = 0; row < NREL * CCH; row++)
        acc += s_ctx[row] * W_rs[(CCH + row) * CCH + t];
    size_t oi = ((size_t)b * CCH + t) * FF + f;
    out[oi] = out[oi] + acc;
}

// ---------------- host launcher ----------------
extern "C" void kernel_launch(void* const* d_in, const int* in_sizes, int n_in,
                              void* d_out, int out_size, void* d_ws, size_t ws_size,
                              hipStream_t stream) {
    (void)in_sizes; (void)n_in; (void)out_size; (void)ws_size;
    const float* input = (const float*)d_in[0];
    const int*   masks = (const int*)  d_in[1];
    const float* W_r1 = (const float*)d_in[2];  const float* b_r1 = (const float*)d_in[3];
    const float* g_r1 = (const float*)d_in[4];  const float* be_r1 = (const float*)d_in[5];
    const float* W_r2 = (const float*)d_in[6];  const float* b_r2 = (const float*)d_in[7];
    const float* W_p1 = (const float*)d_in[8];  const float* b_p1 = (const float*)d_in[9];
    const float* g_p1 = (const float*)d_in[10]; const float* be_p1 = (const float*)d_in[11];
    const float* W_p2 = (const float*)d_in[12]; const float* b_p2 = (const float*)d_in[13];
    const float* g_p2 = (const float*)d_in[14]; const float* be_p2 = (const float*)d_in[15];
    const float* W_p3 = (const float*)d_in[16]; const float* b_p3 = (const float*)d_in[17];
    const float* W_rs = (const float*)d_in[18]; const float* b_rs = (const float*)d_in[19];

    float* out      = (float*)d_out;                               // [BS,C,F]
    float* P_logits = out + (size_t)BSZ * CCH * FF;                // [BS,N*K,3]
    int*   topk_idx = (int*)(P_logits + (size_t)BSZ * NVOX * KSEL * 3);

    float* pe_m = (float*)d_ws;                                    // BS*N*C
    float* xi   = pe_m + (size_t)BSZ * NVOX * CCH;
    float* a_ws = xi   + (size_t)BSZ * NVOX * CCH;                 // BS*N*32
    float* b_ws = a_ws + (size_t)BSZ * NVOX * 32;
    float* topv = b_ws + (size_t)BSZ * NVOX * 32;                  // BS*N*K
    float* ctx  = topv + (size_t)BSZ * NVOX * KSEL;                // BS*N*3*C
    _Float16* wb1 = (_Float16*)(ctx + (size_t)BSZ * NVOX * NREL * CCH);
    _Float16* wb2 = wb1 + 8 * 512;

    hipLaunchKernelGGL(k_pexi, dim3((BSZ * NVOX * CCH) / 256), dim3(256), 0, stream,
                       input, masks, pe_m, xi);
    hipLaunchKernelGGL(k_wfrag, dim3(1), dim3(256), 0, stream, W_p1, W_p2, wb1, wb2);
    hipLaunchKernelGGL(k_ab, dim3((BSZ * NVOX * 32) / 256), dim3(256), 0, stream,
                       xi, pe_m, W_r1, b_r1, a_ws, b_ws);
    hipLaunchKernelGGL(k_relate_topk, dim3(BSZ * NVOX), dim3(256), 0, stream,
                       a_ws, b_ws, g_r1, be_r1, W_r2, b_r2, topv, topk_idx);
    hipLaunchKernelGGL(k_pairmlp, dim3(BSZ * NVOX), dim3(256), 0, stream,
                       xi, pe_m, topk_idx, topv, wb1, wb2,
                       g_p1, b_p1, be_p1, g_p2, b_p2, be_p2, W_p3, b_p3,
                       P_logits, ctx);
    hipLaunchKernelGGL(k_conv_base, dim3((FF + 63) / 64, BSZ), dim3(256), 0, stream,
                       input, W_rs, b_rs, out);
    hipLaunchKernelGGL(k_conv_ctx, dim3(BSZ * NVOX), dim3(128), 0, stream,
                       masks, ctx, W_rs, out);
}
